// LlamaAttention_64321430225461
// MI455X (gfx1250) — matrix operand order repaired and checked by reading, not compiled, not run on hardware
//
#include <hip/hip_runtime.h>

typedef __attribute__((ext_vector_type(16))) __bf16 v16bf;
typedef __attribute__((ext_vector_type(8)))  float  v8f;

#define WMMA_BF16(A_, B_, C_) \
  __builtin_amdgcn_wmma_f32_16x16x32_bf16(false, (A_), false, (B_), (short)0, (C_), false, false)

// LDS byte address of a __shared__ object: low 32 bits of its generic pointer
// (aperture scheme: generic shared addr = {SHARED_BASE_HI, lds_offset}).
#define LDS_ADDR(p) ((unsigned)(uintptr_t)(p))

// CDNA5 async memory->LDS copy, 16B per lane; INST_OFFSET applies to BOTH the
// global and LDS addresses, so one register pair covers offset:0 and offset:16.
#define ASYNC_LD_B128(ldsoff, gaddr)                                         \
  asm volatile("global_load_async_to_lds_b128 %0, %1, off"                   \
               :: "v"(ldsoff), "v"(gaddr) : "memory")
#define ASYNC_LD_B128_OFS16(ldsoff, gaddr)                                   \
  asm volatile("global_load_async_to_lds_b128 %0, %1, off offset:16"         \
               :: "v"(ldsoff), "v"(gaddr) : "memory")
#define WAIT_ASYNC0() asm volatile("s_wait_asynccnt 0x0" ::: "memory")

constexpr int S   = 2048;
constexpr int D   = 4096;
constexpr int NH  = 32;
constexpr int NKV = 8;
constexpr int HD  = 128;
constexpr int E   = D + 2 * NKV * HD;  // 6144 fused QKV width
constexpr int REP = NH / NKV;          // 4

// ---------------------------------------------------------------------------
// fp32 -> bf16 conversion (grid-stride)
// ---------------------------------------------------------------------------
__global__ __launch_bounds__(256)
void cvt_f32_bf16(const float* __restrict__ in, __bf16* __restrict__ out, long n) {
  long i = (long)blockIdx.x * blockDim.x + threadIdx.x;
  long stride = (long)gridDim.x * blockDim.x;
  for (; i < n; i += stride) out[i] = (__bf16)in[i];
}

// ---------------------------------------------------------------------------
// NT GEMM: C[m,n] = sum_k A[m,k] * B[n,k]   (A: MxK row-major, B: NxK row-major)
// block = 128 threads = 4 waves; block tile 64(M) x 64(N); K step 32.
// LDS staging via GLOBAL_LOAD_ASYNC_TO_LDS_B128 (ASYNCcnt), no VGPR transit.
// Each wave owns a 16x64 strip -> 4 x v_wmma_f32_16x16x32_bf16 per K step.
// ---------------------------------------------------------------------------
__global__ __launch_bounds__(128)
void gemm_bf16_nt(const __bf16* __restrict__ A, const __bf16* __restrict__ Bm,
                  float* __restrict__ C, int M, int N, int K) {
  // single shared object; 40-half row pitch (80B) avoids bank conflicts and
  // keeps 16B alignment for B128 transfers at half-offsets 0/8/16/24
  __shared__ __align__(16) __bf16 lA[64][40];
  __shared__ __align__(16) __bf16 lB[64][40];

  const int tid    = threadIdx.x;
  const int lane   = tid & 31;
  const int wid    = tid >> 5;       // 0..3 -> M sub-tile
  const int l16    = lane & 15;
  const int halfId = lane >> 4;      // 0 or 1

  const int m0 = blockIdx.y * 64;
  const int n0 = blockIdx.x * 64;

  // this thread's slice of the cooperative tile load: row r, halves h..h+15
  const int r = tid >> 1;            // 0..63
  const int h = (tid & 1) * 16;      // 0 or 16
  const __bf16* ga = A  + (long)(m0 + r) * K + h;
  const __bf16* gb = Bm + (long)(n0 + r) * K + h;
  const unsigned la = LDS_ADDR(&lA[r][h]);
  const unsigned lb = LDS_ADDR(&lB[r][h]);

  v8f acc[4] = {};

  for (int k0 = 0; k0 < K; k0 += 32) {
    // async memory -> LDS (4 x B128 per thread = 8KB block tile)
    ASYNC_LD_B128(la, ga + k0);
    ASYNC_LD_B128_OFS16(la, ga + k0);
    ASYNC_LD_B128(lb, gb + k0);
    ASYNC_LD_B128_OFS16(lb, gb + k0);
    // prefetch next K tile into cache while WMMAs run on this one
    __builtin_prefetch(ga + k0 + 32, 0, 1);
    __builtin_prefetch(gb + k0 + 32, 0, 1);
    WAIT_ASYNC0();
    __syncthreads();

    // A fragment: lane half 0 holds k 0..15, half 1 holds 16..31
    v16bf afrag;
    {
      const int row = wid * 16 + l16;
      const int kb  = halfId * 16;
      #pragma unroll
      for (int e = 0; e < 8; ++e) afrag[e]     = lA[row][kb + e];
      #pragma unroll
      for (int e = 0; e < 8; ++e) afrag[8 + e] = lA[row][kb + 8 + e];
    }
    // B fragments: n = c*16 + l16 ; k = halfId*16 + e
    #pragma unroll
    for (int c = 0; c < 4; ++c) {
      v16bf bfrag;
      const int n  = c * 16 + l16;
      const int kb = halfId * 16;
      #pragma unroll
      for (int e = 0; e < 16; ++e) bfrag[e] = lB[n][kb + e];
      acc[c] = WMMA_BF16(afrag, bfrag, acc[c]);
    }
    __syncthreads();  // protect LDS from next iteration's async writes
  }

  // C layout: VGPR r -> M = r + 8*halfId ; N = l16
  const int mrow = m0 + wid * 16 + halfId * 8;
  #pragma unroll
  for (int c = 0; c < 4; ++c)
    #pragma unroll
    for (int rr = 0; rr < 8; ++rr)
      C[(long)(mrow + rr) * N + n0 + c * 16 + l16] = acc[c][rr];
}

// ---------------------------------------------------------------------------
// RoPE + GQA scatter: qkv fp32 [S, E] -> Q[NH][S][HD], K[NKV][S][HD], V[NKV][S][HD] bf16
// ---------------------------------------------------------------------------
__global__ __launch_bounds__(128)
void rope_scatter(const float* __restrict__ qkv,
                  __bf16* __restrict__ Q, __bf16* __restrict__ K,
                  __bf16* __restrict__ V) {
  const int s = blockIdx.x;
  const int h = blockIdx.y;    // 0..NH+2*NKV-1
  const int i = threadIdx.x;   // 0..127

  if (h < NH + NKV) {          // Q or K head: apply RoPE
    const int  isQ  = (h < NH);
    const int  hh   = isQ ? h : (h - NH);
    const long base = (long)s * E + (isQ ? hh * HD : D + hh * HD);
    const float u   = qkv[base + i];
    const int   j   = i & 63;                                // i % (HD/2)
    const float ang = (float)s * __powf(10000.0f, -(float)j * (1.0f / 64.0f));
    const float cs  = __cosf(ang);
    const float sn  = __sinf(ang);
    const float rot = (i < 64) ? -qkv[base + i + 64] : qkv[base + i - 64];
    const __bf16 o  = (__bf16)(u * cs + rot * sn);
    if (isQ) Q[((long)hh * S + s) * HD + i] = o;
    else     K[((long)hh * S + s) * HD + i] = o;
  } else {                      // V head: straight convert
    const int vh = h - NH - NKV;
    V[((long)vh * S + s) * HD + i] =
        (__bf16)qkv[(long)s * E + D + NKV * HD + vh * HD + i];
  }
}

// ---------------------------------------------------------------------------
// Flash attention: 1 wave per (head, 16-query tile). Online softmax.
// scores: 4 WMMA per 16x16 tile (contraction over HD=128), two tiles / 32-key step
// attn*V: 8 WMMA per 32-key step (contraction over keys)
// ---------------------------------------------------------------------------
__global__ __launch_bounds__(32)
void flash_attn(const __bf16* __restrict__ Q, const __bf16* __restrict__ K,
                const __bf16* __restrict__ V, __bf16* __restrict__ Y) {
  const int q0   = blockIdx.x * 16;
  const int head = blockIdx.y;
  const int kvh  = head / REP;

  const int lane   = threadIdx.x;
  const int l16    = lane & 15;
  const int halfId = lane >> 4;

  __shared__ float sP[16][36];   // P staging for C->A fragment re-layout

  // Load the 16x128 Q tile as 4 A-fragments (k chunk of 32 each)
  v16bf qf[4];
  {
    const __bf16* qbase = Q + ((long)head * S + q0) * HD;
    #pragma unroll
    for (int c = 0; c < 4; ++c) {
      const long rowoff = (long)l16 * HD + c * 32 + halfId * 16;
      #pragma unroll
      for (int e = 0; e < 8; ++e) qf[c][e]     = qbase[rowoff + e];
      #pragma unroll
      for (int e = 0; e < 8; ++e) qf[c][8 + e] = qbase[rowoff + 8 + e];
    }
  }

  v8f O[8] = {};
  float mrow[8], lrow[8], alpha[8];
  #pragma unroll
  for (int r = 0; r < 8; ++r) { mrow[r] = -3.0e38f; lrow[r] = 0.0f; }

  const float scale = 0.08838834764831845f;  // 1/sqrt(128)
  const int kend = q0 + 16;                  // causal bound (exclusive)

  for (int kb0 = 0; kb0 < kend; kb0 += 32) {
    // ---- scores: two 16x16 tiles (keys kb0..+15, kb0+16..+31) ----
    v8f sc[2];
    #pragma unroll
    for (int t = 0; t < 2; ++t) {
      v8f s = {};
      const int n0 = kb0 + t * 16;
      #pragma unroll
      for (int c = 0; c < 4; ++c) {
        v16bf kf;
        const __bf16* kp =
            K + ((long)kvh * S + n0 + l16) * HD + c * 32 + halfId * 16;
        #pragma unroll
        for (int e = 0; e < 16; ++e) kf[e] = kp[e];
        s = WMMA_BF16(qf[c], kf, s);
      }
      sc[t] = s;
    }

    // ---- scale + causal mask + online softmax update ----
    #pragma unroll
    for (int r = 0; r < 8; ++r) {
      const int qrow = q0 + r + 8 * halfId;
      float v0 = sc[0][r] * scale;
      float v1 = sc[1][r] * scale;
      if (kb0 + l16      > qrow) v0 = -3.0e38f;
      if (kb0 + 16 + l16 > qrow) v1 = -3.0e38f;

      float mx = fmaxf(v0, v1);
      #pragma unroll
      for (int off = 1; off < 16; off <<= 1)
        mx = fmaxf(mx, __shfl_xor(mx, off, 32));   // reduce within 16-lane half
      const float mnew = fmaxf(mrow[r], mx);
      const float a    = __expf(mrow[r] - mnew);
      const float p0   = __expf(v0 - mnew);
      const float p1   = __expf(v1 - mnew);
      float ps = p0 + p1;
      #pragma unroll
      for (int off = 1; off < 16; off <<= 1)
        ps += __shfl_xor(ps, off, 32);
      lrow[r]  = lrow[r] * a + ps;
      mrow[r]  = mnew;
      alpha[r] = a;
      sc[0][r] = p0;
      sc[1][r] = p1;
    }

    // rescale running output
    #pragma unroll
    for (int c = 0; c < 8; ++c)
      #pragma unroll
      for (int r = 0; r < 8; ++r) O[c][r] *= alpha[r];

    // ---- re-layout P (C layout) -> A fragment via LDS ----
    #pragma unroll
    for (int r = 0; r < 8; ++r) {
      sP[r + 8 * halfId][l16]      = sc[0][r];
      sP[r + 8 * halfId][16 + l16] = sc[1][r];
    }
    __syncthreads();
    v16bf pf;
    {
      const int kb = halfId * 16;
      #pragma unroll
      for (int e = 0; e < 8; ++e) pf[e]     = (__bf16)sP[l16][kb + e];
      #pragma unroll
      for (int e = 0; e < 8; ++e) pf[8 + e] = (__bf16)sP[l16][kb + 8 + e];
    }

    // ---- O += P(16x32) x V(32x16) per hd chunk ----
    #pragma unroll
    for (int c = 0; c < 8; ++c) {
      v16bf vf;
      const __bf16* vp =
          V + ((long)kvh * S + kb0 + halfId * 16) * HD + c * 16 + l16;
      #pragma unroll
      for (int e = 0; e < 16; ++e) vf[e] = vp[(long)e * HD];
      O[c] = WMMA_BF16(pf, vf, O[c]);
    }
    __syncthreads();
  }

  // ---- normalize and store Y[s, head*HD + hd] (bf16 for final projection) ----
  #pragma unroll
  for (int c = 0; c < 8; ++c)
    #pragma unroll
    for (int r = 0; r < 8; ++r) {
      const int srow = q0 + r + 8 * halfId;
      const float o  = O[c][r] / lrow[r];
      Y[(long)srow * (NH * HD) + head * HD + c * 16 + l16] = (__bf16)o;
    }
}

// ---------------------------------------------------------------------------
// Launch
// ---------------------------------------------------------------------------
extern "C" void kernel_launch(void* const* d_in, const int* in_sizes, int n_in,
                              void* d_out, int out_size, void* d_ws, size_t ws_size,
                              hipStream_t stream) {
  const float* x    = (const float*)d_in[0];
  // d_in[1] = mask (causal, computed analytically in-kernel)
  const float* wqkv = (const float*)d_in[2];
  const float* wo   = (const float*)d_in[3];
  float* out = (float*)d_out;

  char* ws = (char*)d_ws;
  __bf16* xb    = (__bf16*)ws;  ws += (size_t)S * D * sizeof(__bf16);
  __bf16* wqkvb = (__bf16*)ws;  ws += (size_t)E * D * sizeof(__bf16);
  __bf16* wob   = (__bf16*)ws;  ws += (size_t)D * D * sizeof(__bf16);
  float*  qkvf  = (float*) ws;  ws += (size_t)S * E * sizeof(float);
  __bf16* Qb    = (__bf16*)ws;  ws += (size_t)NH  * S * HD * sizeof(__bf16);
  __bf16* Kb    = (__bf16*)ws;  ws += (size_t)NKV * S * HD * sizeof(__bf16);
  __bf16* Vb    = (__bf16*)ws;  ws += (size_t)NKV * S * HD * sizeof(__bf16);
  __bf16* Yb    = (__bf16*)ws;  ws += (size_t)S * D * sizeof(__bf16);

  // 1) precision conversion (bf16 inputs for WMMA, fp32 accumulate)
  cvt_f32_bf16<<<2048, 256, 0, stream>>>(x,    xb,    (long)S * D);
  cvt_f32_bf16<<<4096, 256, 0, stream>>>(wqkv, wqkvb, (long)E * D);
  cvt_f32_bf16<<<2048, 256, 0, stream>>>(wo,   wob,   (long)D * D);

  // 2) fused QKV projection: qkv[s,e] = sum_d x[s,d] * wqkv[e,d]
  {
    dim3 grid(E / 64, S / 64);
    gemm_bf16_nt<<<grid, 128, 0, stream>>>(xb, wqkvb, qkvf, S, E, D);
  }

  // 3) RoPE + head-major scatter (GQA layout)
  {
    dim3 grid(S, NH + 2 * NKV);
    rope_scatter<<<grid, 128, 0, stream>>>(qkvf, Qb, Kb, Vb);
  }

  // 4) causal flash attention
  {
    dim3 grid(S / 16, NH);
    flash_attn<<<grid, 32, 0, stream>>>(Qb, Kb, Vb, Yb);
  }

  // 5) output projection: out[s,e] = sum_d y[s,d] * wo[e,d]
  {
    dim3 grid(D / 64, S / 64);
    gemm_bf16_nt<<<grid, 128, 0, stream>>>(Yb, wob, out, S, D, D);
  }
}
